// Joint_16784732193240
// MI455X (gfx1250) — matrix operand order repaired and checked by reading, not compiled, not run on hardware
//
#include <hip/hip_runtime.h>
#include <hip/hip_bf16.h>

// ---------------- problem constants ----------------
#define B_DIM   4
#define T_DIM   200
#define U_DIM   100
#define D_MOD   512
#define D_HID   1024
#define D_OUT   512

// ---------------- main-kernel tile constants ----------------
#define TT 16              // t rows per M-tile
#define UT 8               // u rows per M-tile
#define MT 128             // TT*UT rows per workgroup
#define NT 256             // output columns per workgroup
#define KT 32              // K per step (WMMA bf16 K)
#define KSTEPS (D_HID / KT)
#define LDSS 40            // LDS row stride in ushorts (32 data + 8 pad -> 80B, conflict-free b128)
#define TTILES 13          // ceil(200/16)
#define UTILES 13          // ceil(100/8)

#define AS1 __attribute__((address_space(1)))
#define AS3 __attribute__((address_space(3)))

// matches the builtin's parameter pointee type exactly
typedef int v4i_t __attribute__((__vector_size__(4 * sizeof(int))));

// CDNA5 async global->LDS path (ASYNCcnt), guarded so either toolchain compiles
#if defined(__has_builtin)
# if __has_builtin(__builtin_amdgcn_global_load_async_to_lds_b128)
#  define HAVE_ASYNC_LDS 1
# endif
#endif
#ifndef HAVE_ASYNC_LDS
# define HAVE_ASYNC_LDS 0
#endif

#if defined(__has_builtin)
# if __has_builtin(__builtin_amdgcn_s_wait_asynccnt)
#  define WAIT_ASYNC0() __builtin_amdgcn_s_wait_asynccnt(0)
# else
#  define WAIT_ASYNC0() asm volatile("s_wait_asynccnt 0x0" ::: "memory")
# endif
#else
# define WAIT_ASYNC0() asm volatile("s_wait_asynccnt 0x0" ::: "memory")
#endif

typedef __bf16 bf16x8  __attribute__((ext_vector_type(8)));
typedef __bf16 bf16x16 __attribute__((ext_vector_type(16)));
typedef float  f32x4   __attribute__((ext_vector_type(4)));
typedef float  f32x8   __attribute__((ext_vector_type(8)));
typedef unsigned short u16x4 __attribute__((ext_vector_type(4)));
typedef unsigned short u16x8 __attribute__((ext_vector_type(8)));

__device__ __forceinline__ int imin(int a, int b) { return a < b ? a : b; }

// fp32 -> bf16 round-to-nearest-even
__device__ __forceinline__ unsigned short f2bf(float f) {
  unsigned u = __builtin_bit_cast(unsigned, f);
  unsigned r = u + 0x7FFFu + ((u >> 16) & 1u);
  return (unsigned short)(r >> 16);
}

// tanh(x) = 1 - 2/(exp(2x)+1), via v_exp_f32 + v_rcp_f32 (correct limits at +-inf)
__device__ __forceinline__ float fast_tanh(float x) {
  float e2x = __builtin_amdgcn_exp2f(x * 2.885390081777927f); // 2*log2(e)
  return 1.0f - 2.0f * __builtin_amdgcn_rcpf(e2x + 1.0f);
}

// ---------------------------------------------------------------------------
// Kernel 0: W2 fp32 -> bf16 (O=512 rows x H=1024, row-major, H contiguous)
// ---------------------------------------------------------------------------
__global__ __launch_bounds__(256) void cvt_w2(const float* __restrict__ W2,
                                              unsigned short* __restrict__ w2bf) {
  int i = (blockIdx.x * 256 + threadIdx.x) * 4;
  if (i < D_OUT * D_HID) {
    f32x4 v = *(const f32x4*)(W2 + i);
    u16x4 o;
    o[0] = f2bf(v[0]); o[1] = f2bf(v[1]); o[2] = f2bf(v[2]); o[3] = f2bf(v[3]);
    *(u16x4*)(w2bf + i) = o;
  }
}

// ---------------------------------------------------------------------------
// Kernel 1: enc_h = enc @ W1[:, :512]^T + b1   (rows = B*T = 800)
//           dec_h = dec @ W1[:, 512:]^T        (rows = B*U = 400)
// fp32 precision preserved pre-tanh. W1 16-row slice staged in LDS.
// ---------------------------------------------------------------------------
__global__ __launch_bounds__(256) void hidden_gemm(const float* __restrict__ enc,
                                                   const float* __restrict__ dec,
                                                   const float* __restrict__ W1,
                                                   const float* __restrict__ b1,
                                                   float* __restrict__ ench,
                                                   float* __restrict__ dech) {
  __shared__ __align__(16) float W1s[16 * 516];   // pad 512->516 floats: conflict-free
  const int hb = blockIdx.x;        // 16 h-columns per block
  const int rb = blockIdx.y;        // 128 rows per block
  const int z  = blockIdx.z;        // 0 = enc, 1 = dec
  const int nrows = z ? (B_DIM * U_DIM) : (B_DIM * T_DIM);
  if (z && rb >= 4) return;         // uniform across block

  const int tid = threadIdx.x;
  const int h0 = hb * 16;

  for (int i = tid; i < 16 * 128; i += 256) {
    int rr = i >> 7;
    int cc = (i & 127) << 2;
    f32x4 v = *(const f32x4*)(W1 + (size_t)(h0 + rr) * 1024 + z * 512 + cc);
    *(f32x4*)&W1s[rr * 516 + cc] = v;
  }
  __syncthreads();

  const int c    = tid & 15;
  const int rgrp = tid >> 4;
  const float* x    = z ? dec  : enc;
  float*       outh = z ? dech : ench;
  const float  bias = z ? 0.0f : b1[h0 + c];

  #pragma unroll
  for (int it = 0; it < 8; ++it) {
    int rr = rb * 128 + it * 16 + rgrp;
    if (rr < nrows) {
      const float* xr = x + (size_t)rr * D_MOD;
      const float* wr = &W1s[c * 516];
      float a0 = 0.f, a1 = 0.f, a2 = 0.f, a3 = 0.f;
      #pragma unroll 4
      for (int k = 0; k < D_MOD; k += 4) {
        f32x4 xv = *(const f32x4*)(xr + k);
        f32x4 wv = *(const f32x4*)(wr + k);
        a0 = fmaf(xv[0], wv[0], a0);
        a1 = fmaf(xv[1], wv[1], a1);
        a2 = fmaf(xv[2], wv[2], a2);
        a3 = fmaf(xv[3], wv[3], a3);
      }
      outh[(size_t)rr * D_HID + h0 + c] = (a0 + a1) + (a2 + a3) + bias;
    }
  }
}

// ---------------------------------------------------------------------------
// Kernel 2: fused  out[b,t,u,:] = tanh(enc_h[b,t,:] + dec_h[b,u,:]) @ W2bf^T + b2
// Double-buffered software pipeline, one barrier per K-step:
//   top:  [Bt waves] s_wait_asynccnt 0  ->  barrier (tile ks published)
//   then: issue next-tile fills (async B-tile to LDS; prefetch enc/dec rows)
//   then: 16x v_wmma_f32_16x16x32_bf16 on tile ks (B-frags batched 4-wide)
//   then: tanh-produce next hA tile (loads hit prefetched lines)
// ---------------------------------------------------------------------------
__global__ __launch_bounds__(256) void joint_main(const float* __restrict__ ench,
                                                  const float* __restrict__ dech,
                                                  const unsigned short* __restrict__ w2bf,
                                                  const float* __restrict__ b2,
                                                  float* __restrict__ out) {
  __shared__ __align__(16) unsigned short hA[2][MT * LDSS];  // 2 x 10240 B
  __shared__ __align__(16) unsigned short Bt[2][NT * LDSS];  // 2 x 20480 B

  const int nt = blockIdx.x;                 // 0..1
  const int mt = blockIdx.y;                 // 0..168
  const int b  = blockIdx.z;                 // 0..3
  const int tt = mt / UTILES, ut = mt % UTILES;
  const int t0 = tt * TT, u0 = ut * UT, n0 = nt * NT;

  const int tid  = threadIdx.x;
  const int wave = tid >> 5;
  const int lane = tid & 31;
  const int mw = wave >> 1;                  // 0..3  (M position of wave)
  const int nw = wave & 1;                   // 0..1  (N position of wave)
  const int rsel  = lane & 15;
  const int khalf = lane >> 4;               // 0 or 1

  f32x8 acc[2][8] = {};                      // 32(M) x 128(N) per wave

  // hA producer setup (waves 0-3; clamped reads keep addresses valid + EXEC full)
  const int mrow = tid & 127;
  const int tcl = imin(t0 + (mrow >> 3), T_DIM - 1);
  const int ucl = imin(u0 + (mrow & 7), U_DIM - 1);
  const float* erow = ench + (size_t)(b * T_DIM + tcl) * D_HID;
  const float* drow = dech + (size_t)(b * U_DIM + ucl) * D_HID;

  // Bt producer setup (waves 4-7; thread handles rows 2*btid, 2*btid+1)
  const int btid = tid & 127;
  const unsigned short* wrow0 = w2bf + (size_t)(n0 + 2 * btid) * D_HID;
  const unsigned short* wrow1 = wrow0 + D_HID;

  auto produce_hA = [&](int k0, int buf) {
    const float* ep = erow + k0;
    const float* dp = drow + k0;
    u16x8 pk[4];
    #pragma unroll
    for (int i = 0; i < 8; ++i) {
      f32x4 ev = *(const f32x4*)(ep + 4 * i);
      f32x4 dv = *(const f32x4*)(dp + 4 * i);
      #pragma unroll
      for (int c = 0; c < 4; ++c) {
        int e = 4 * i + c;
        pk[e >> 3][e & 7] = f2bf(fast_tanh(ev[c] + dv[c]));
      }
    }
    u16x8* dst = (u16x8*)&hA[buf][mrow * LDSS];
    #pragma unroll
    for (int c = 0; c < 4; ++c) dst[c] = pk[c];
  };

  auto produce_Bt = [&](int k0, int buf) {
#if HAVE_ASYNC_LDS
    #pragma unroll
    for (int rr = 0; rr < 2; ++rr) {
      const unsigned short* src = (rr ? wrow1 : wrow0) + k0;
      unsigned short* dstrow = &Bt[buf][(2 * btid + rr) * LDSS];
      #pragma unroll
      for (int c = 0; c < 4; ++c) {
        __builtin_amdgcn_global_load_async_to_lds_b128(
            (AS1 v4i_t*)(src + 8 * c), (AS3 v4i_t*)(dstrow + 8 * c), 0, 0);
      }
    }
#else
    u16x8 w[8];
    #pragma unroll
    for (int rr = 0; rr < 2; ++rr) {
      const unsigned short* src = (rr ? wrow1 : wrow0) + k0;
      #pragma unroll
      for (int c = 0; c < 4; ++c) w[rr * 4 + c] = ((const u16x8*)src)[c];
    }
    #pragma unroll
    for (int rr = 0; rr < 2; ++rr) {
      u16x8* dstrow = (u16x8*)&Bt[buf][(2 * btid + rr) * LDSS];
      #pragma unroll
      for (int c = 0; c < 4; ++c) dstrow[c] = w[rr * 4 + c];
    }
#endif
  };

  // ---- prologue: fill tile 0 into buffer 0 ----
  if (tid < 128) produce_hA(0, 0);
  else           produce_Bt(0, 0);

  #pragma unroll 1
  for (int ks = 0; ks < KSTEPS; ++ks) {
    const int cur = ks & 1, nxt = cur ^ 1;
    const int kn = (ks + 1) * KT;
    const bool has_next = (ks + 1) < KSTEPS;

#if HAVE_ASYNC_LDS
    if (tid >= 128) WAIT_ASYNC0();           // tile ks B-data committed to LDS
#endif
    __syncthreads();                         // tile ks visible; prior reads of buf nxt done

    // issue next-tile fills early: overlap with WMMAs below
    if (has_next) {
      if (tid < 128) {
        __builtin_prefetch(erow + kn, 0, 3); // global_prefetch_b8 (cacheline covers 128B chunk)
        __builtin_prefetch(drow + kn, 0, 3);
      }
#if HAVE_ASYNC_LDS
      else {
        produce_Bt(kn, nxt);                 // ASYNCcnt-tracked, no VGPR data
      }
#endif
    }

    // --- consume tile ks ---
    // A frags (ISA 16-bit A layout: K {0..7,16..23} lanes 0-15 / {8..15,24..31} lanes 16-31)
    bf16x16 afrag[2];
    #pragma unroll
    for (int g = 0; g < 2; ++g) {
      int row = mw * 32 + g * 16 + rsel;
      const bf16x8* p = (const bf16x8*)&hA[cur][row * LDSS + khalf * 8];
      afrag[g] = __builtin_shufflevector(p[0], p[2],
                   0,1,2,3,4,5,6,7,8,9,10,11,12,13,14,15);
    }
    // B frags (K {0..7,16..23} lanes 0-15 / {8..15,24..31} lanes 16-31), batched 4-wide
    #pragma unroll
    for (int jj = 0; jj < 8; jj += 4) {
      bf16x16 bf[4];
      #pragma unroll
      for (int j2 = 0; j2 < 4; ++j2) {
        int nrow = nw * 128 + (jj + j2) * 16 + rsel;
        const bf16x8* q = (const bf16x8*)&Bt[cur][nrow * LDSS + khalf * 8];
        bf[j2] = __builtin_shufflevector(q[0], q[2],
                   0,1,2,3,4,5,6,7,8,9,10,11,12,13,14,15);
      }
      #pragma unroll
      for (int j2 = 0; j2 < 4; ++j2) {
        acc[0][jj + j2] = __builtin_amdgcn_wmma_f32_16x16x32_bf16(
            false, afrag[0], false, bf[j2], (short)0, acc[0][jj + j2], false, false);
        acc[1][jj + j2] = __builtin_amdgcn_wmma_f32_16x16x32_bf16(
            false, afrag[1], false, bf[j2], (short)0, acc[1][jj + j2], false, false);
      }
    }

    // --- produce next tile (hA tanh; Bt too when no async path) ---
    if (has_next) {
      if (tid < 128) produce_hA(kn, nxt);
#if !HAVE_ASYNC_LDS
      else           produce_Bt(kn, nxt);
#endif
    }
  }

  // --- epilogue: D layout (VGPR r -> M = 8*khalf + r, N = rsel), +b2, guarded store ---
  const size_t outBase = (size_t)b * T_DIM * U_DIM * D_OUT;
  #pragma unroll
  for (int g = 0; g < 2; ++g) {
    #pragma unroll
    for (int j = 0; j < 8; ++j) {
      int col = n0 + nw * 128 + j * 16 + rsel;
      float bb = b2[col];
      #pragma unroll
      for (int r = 0; r < 8; ++r) {
        int mr = mw * 32 + g * 16 + khalf * 8 + r;
        int t = t0 + (mr >> 3);
        int u = u0 + (mr & 7);
        if (t < T_DIM && u < U_DIM) {
          out[outBase + ((size_t)t * U_DIM + u) * D_OUT + col] = acc[g][j][r] + bb;
        }
      }
    }
  }
}

// ---------------------------------------------------------------------------
extern "C" void kernel_launch(void* const* d_in, const int* in_sizes, int n_in,
                              void* d_out, int out_size, void* d_ws, size_t ws_size,
                              hipStream_t stream) {
  const float* enc = (const float*)d_in[0];   // (4,200,512)
  const float* dec = (const float*)d_in[1];   // (4,100,512)
  const float* W1  = (const float*)d_in[2];   // (1024,1024)
  const float* b1  = (const float*)d_in[3];   // (1024,)
  const float* W2  = (const float*)d_in[4];   // (512,1024)
  const float* b2  = (const float*)d_in[5];   // (512,)
  float* out = (float*)d_out;                 // (4,200,100,512)

  // workspace layout (~5.97 MB): enc_h fp32 | dec_h fp32 | W2 bf16
  float* ench = (float*)d_ws;                             // 800*1024 f32
  float* dech = ench + (size_t)B_DIM * T_DIM * D_HID;     // 400*1024 f32
  unsigned short* w2bf = (unsigned short*)(dech + (size_t)B_DIM * U_DIM * D_HID);

  cvt_w2<<<(D_OUT * D_HID) / (256 * 4), 256, 0, stream>>>(W2, w2bf);
  hidden_gemm<<<dim3(D_HID / 16, 7, 2), 256, 0, stream>>>(enc, dec, W1, b1, ench, dech);
  joint_main<<<dim3(D_OUT / NT, TTILES * UTILES, B_DIM), 256, 0, stream>>>(
      ench, dech, w2bf, b2, out);
}
